// GraphEncoderGPS_12927851561631
// MI455X (gfx1250) — matrix operand order repaired and checked by reading, not compiled, not run on hardware
//
#include <hip/hip_runtime.h>
#include <hip/hip_bf16.h>

// ---------------------------------------------------------------------------
// GraphEncoderGPS for MI455X (gfx1250, wave32, WMMA + TDM).
// Dense GEMMs: f16 operands (converted once), f32 accumulate on
// v_wmma_f32_16x16x32_f16; A tiles staged to LDS via tensor_load_to_lds.
// ---------------------------------------------------------------------------

typedef __attribute__((ext_vector_type(16))) _Float16     v16h;
typedef __attribute__((ext_vector_type(8)))  float        v8f;
typedef __attribute__((ext_vector_type(4)))  unsigned int v4u;
typedef __attribute__((ext_vector_type(8)))  int          v8i;
typedef __attribute__((ext_vector_type(4)))  int          v4i;

constexpr int NB   = 32;     // batch
constexpr int NN   = 512;    // nodes
constexpr int NE   = 4096;   // edges
constexpr int ND   = 256;    // model dim
constexpr int NH   = 8;      // heads
constexpr int NHD  = 32;     // head dim
constexpr int NDFF = 2048;   // ffn dim
constexpr int NR   = 16;     // relations
constexpr int NL   = 3;      // layers
constexpr int NK   = 256;    // top-k
constexpr float LN_EPS = 1e-5f;

// ---------------------------------------------------------------------------
// WMMA GEMM (f16 in, f32 out):  C[m,n] = sum_k A[m,k]*W[n,k] (+bias[n]) (relu)
//   A: (M x Kd) f16 row-major, W: (Nc x Kd) f16 row-major (pre-transposed).
//   Block = 128 threads (4 waves); block tile = 128 rows x 16 cols.
//   A tile (128 x 32 halfs, 8KB) staged to LDS by the Tensor Data Mover,
//   issued by wave 0, synchronized with s_wait_tensorcnt + barrier.
// ---------------------------------------------------------------------------
#define GEMM_MT 2   // 16-row tiles per wave

__device__ inline void tdm_load_tile(unsigned lds_addr, const _Float16* gsrc,
                                     int Kd, int Mrem, int lda) {
    const unsigned long long ga = (unsigned long long)(uintptr_t)gsrc;
    v4u g0;
    g0[0] = 1u;                                     // count=1 (valid user D#)
    g0[1] = lds_addr;                               // lds_addr
    g0[2] = (unsigned)(ga & 0xFFFFFFFFu);           // global_addr[31:0]
    g0[3] = (unsigned)((ga >> 32) & 0x01FFFFFFu) |  // global_addr[56:32]
            (2u << 30);                             // type=2 (image)
    v8i g1;
    g1[0] = (int)(1u << 16);                        // data_size=1 (2 bytes)
    g1[1] = (int)(((unsigned)Kd & 0xFFFFu) << 16);              // tensor_dim0 lo
    g1[2] = (int)(((unsigned)Kd >> 16) |
                  (((unsigned)Mrem & 0xFFFFu) << 16));          // dim0 hi|dim1 lo
    g1[3] = (int)((((unsigned)Mrem >> 16) & 0xFFFFu) |
                  (32u << 16));                                 // dim1 hi|tile_dim0=32
    g1[4] = 128;                                    // tile_dim1=128, tile_dim2=0
    g1[5] = lda;                                    // tensor_dim0_stride[31:0]
    g1[6] = 0;                                      // stride hi | dim1_stride lo
    g1[7] = 0;
    v4i gz = {};
#if __clang_major__ >= 23
    v8i gz8 = {};
    __builtin_amdgcn_tensor_load_to_lds(g0, g1, gz, gz, gz8, 0);
#else
    __builtin_amdgcn_tensor_load_to_lds(g0, g1, gz, gz, 0);
#endif
}

__global__ __launch_bounds__(128) void wmma_gemm_f16t(
    const _Float16* __restrict__ A, const _Float16* __restrict__ W,
    const float* __restrict__ bias, float* __restrict__ C,
    int M, int Nc, int Kd, int lda, int ldw, int ldc, int relu)
{
    __shared__ _Float16 atile[128 * 32];            // 8KB staged A tile

    const int tid  = threadIdx.x;
    const int wave = tid >> 5;
    const int lane = tid & 31;
    const int row  = lane & 15;
    const int grp  = lane >> 4;                     // 0 or 1
    const int m0   = blockIdx.x * 128;
    const int n0   = blockIdx.y * 16;
    const int ncol = n0 + row;

    v8f zero = {};
    v8f acc[GEMM_MT];
#pragma unroll
    for (int t = 0; t < GEMM_MT; ++t) acc[t] = zero;

    const unsigned lds_base = (unsigned)(uintptr_t)(void*)atile;

    for (int k0 = 0; k0 < Kd; k0 += 32) {
        __syncthreads();                            // tile consumed
        if (wave == 0) {
            tdm_load_tile(lds_base, A + (long long)m0 * lda + k0,
                          Kd, M, lda);
            __builtin_amdgcn_s_wait_tensorcnt(0);
        }
        __syncthreads();                            // tile ready

        // ---- B fragment: two 8-half runs per lane (k = 8*grp + i, 16 + 8*grp + i)
        const _Float16* pb = W + (long long)ncol * ldw + k0 + grp * 8;
        __builtin_prefetch(pb + 32, 0, 0);          // next K tile
        v16h bf;
#pragma unroll
        for (int i = 0; i < 8; ++i) bf[i] = pb[i];
#pragma unroll
        for (int i = 0; i < 8; ++i) bf[8 + i] = pb[16 + i];

        // ---- A fragments from LDS, 2 stacked M tiles reuse bf
#pragma unroll
        for (int t = 0; t < GEMM_MT; ++t) {
            const _Float16* pa =
                atile + (wave * GEMM_MT * 16 + t * 16 + row) * 32 + grp * 8;
            v16h af;
#pragma unroll
            for (int i = 0; i < 8; ++i) af[i]     = pa[i];
#pragma unroll
            for (int i = 0; i < 8; ++i) af[8 + i] = pa[16 + i];
            acc[t] = __builtin_amdgcn_wmma_f32_16x16x32_f16(
                false, af, false, bf, (short)0, acc[t], false, false);
        }
    }

    const float bval = bias ? bias[ncol] : 0.0f;
#pragma unroll
    for (int t = 0; t < GEMM_MT; ++t) {
#pragma unroll
        for (int i = 0; i < 8; ++i) {
            const int m = m0 + wave * GEMM_MT * 16 + t * 16 + grp * 8 + i;
            float v = acc[t][i] + bval;
            if (relu) v = fmaxf(v, 0.0f);
            C[(long long)m * ldc + ncol] = v;
        }
    }
}

// ---------------------------------------------------------------------------
// Precision conversion kernels (run once per operand; bandwidth-trivial)
// ---------------------------------------------------------------------------
__global__ void cvt_f32_f16(const float* __restrict__ in,
                            _Float16* __restrict__ out, long long n) {
    const long long i = (long long)blockIdx.x * blockDim.x + threadIdx.x;
    if (i < n) out[i] = (_Float16)in[i];
}

// in: (rows_in x cols_in) f32 -> out: (cols_in x rows_in) f16 (transposed)
__global__ void cvt_f32_f16_T(const float* __restrict__ in,
                              _Float16* __restrict__ out,
                              int rows_in, int cols_in) {
    const long long i = (long long)blockIdx.x * blockDim.x + threadIdx.x;
    if (i >= (long long)rows_in * cols_in) return;
    const int n = (int)(i / rows_in);
    const int k = (int)(i % rows_in);
    out[i] = (_Float16)in[(long long)k * cols_in + n];
}

// ---------------------------------------------------------------------------
// Elementwise / graph kernels
// ---------------------------------------------------------------------------
__global__ void zero_kernel(float* p, long long n) {
    const long long i = (long long)blockIdx.x * blockDim.x + threadIdx.x;
    if (i < n) p[i] = 0.0f;
}

__global__ void embed_kernel(const int* __restrict__ ids,
                             const float* __restrict__ emb,
                             float* __restrict__ x) {
    const int bn = blockIdx.x;
    const int d  = threadIdx.x;
    const long long id = ids[bn];
    x[(long long)bn * ND + d] = emb[id * ND + d];
}

__global__ void rgcn_count(const int* __restrict__ ei,
                           const int* __restrict__ et,
                           float* __restrict__ cnt) {
    const int i = blockIdx.x * blockDim.x + threadIdx.x;
    if (i >= NB * NE) return;
    const int b = i / NE, e = i % NE;
    const int dst = ei[b * 2 * NE + NE + e];
    const int r   = et[i];
    atomicAdd(&cnt[(b * NN + dst) * NR + r], 1.0f);
}

__global__ __launch_bounds__(256) void rgcn_scatter(
    const int* __restrict__ ei, const int* __restrict__ et,
    const float* __restrict__ cnt, const float* __restrict__ hr,
    float* __restrict__ out, int r) {
    const int e = blockIdx.x;
    const int b = blockIdx.y;
    if (et[b * NE + e] != r) return;
    const int src = ei[b * 2 * NE + e];
    const int dst = ei[b * 2 * NE + NE + e];
    const float c    = cnt[(b * NN + dst) * NR + r];
    const float norm = 1.0f / fmaxf(c, 1.0f);
    const int d = threadIdx.x;
    atomicAdd(&out[((long long)b * NN + dst) * ND + d],
              hr[((long long)b * NN + src) * ND + d] * norm);
}

__global__ __launch_bounds__(256) void ln_add(
    const float* __restrict__ a, const float* __restrict__ bsrc,
    const float* __restrict__ w, const float* __restrict__ bias,
    float* __restrict__ out) {
    const long long row = blockIdx.x;
    const int t = threadIdx.x;
    __shared__ float red[256];
    float v = a[row * ND + t] + (bsrc ? bsrc[row * ND + t] : 0.0f);
    red[t] = v;
    __syncthreads();
    for (int s = 128; s > 0; s >>= 1) {
        if (t < s) red[t] += red[t + s];
        __syncthreads();
    }
    const float mu = red[0] * (1.0f / ND);
    __syncthreads();
    const float dv = v - mu;
    red[t] = dv * dv;
    __syncthreads();
    for (int s = 128; s > 0; s >>= 1) {
        if (t < s) red[t] += red[t + s];
        __syncthreads();
    }
    const float var = red[0] * (1.0f / ND);
    out[row * ND + t] = dv * rsqrtf(var + LN_EPS) * w[t] + bias[t];
}

__global__ __launch_bounds__(128) void attn_fused(
    const float* __restrict__ qkv, float* __restrict__ o) {
    const int n = blockIdx.x, h = blockIdx.y, b = blockIdx.z;
    const int t = threadIdx.x;
    __shared__ float q[NHD];
    __shared__ float s[NN];
    __shared__ float red[128];
    __shared__ float ored[4][NHD];
    const float* base = qkv + (long long)b * NN * (3 * ND);
    if (t < NHD) q[t] = base[(long long)n * 3 * ND + h * NHD + t];
    __syncthreads();
    const float scale = 0.17677669529663687f;  // 1/sqrt(32)
    for (int j = t; j < NN; j += 128) {
        const float* krow = base + (long long)j * 3 * ND + ND + h * NHD;
        float acc = 0.0f;
#pragma unroll
        for (int i = 0; i < NHD; ++i) acc += q[i] * krow[i];
        s[j] = acc * scale;
    }
    __syncthreads();
    float m = -3.4e38f;
    for (int j = t; j < NN; j += 128) m = fmaxf(m, s[j]);
    red[t] = m;
    __syncthreads();
    for (int st = 64; st > 0; st >>= 1) {
        if (t < st) red[t] = fmaxf(red[t], red[t + st]);
        __syncthreads();
    }
    m = red[0];
    __syncthreads();
    float sum = 0.0f;
    for (int j = t; j < NN; j += 128) {
        const float e = __expf(s[j] - m);
        s[j] = e;
        sum += e;
    }
    red[t] = sum;
    __syncthreads();
    for (int st = 64; st > 0; st >>= 1) {
        if (t < st) red[t] += red[t + st];
        __syncthreads();
    }
    const float inv = 1.0f / red[0];
    __syncthreads();
    const int i = t & (NHD - 1), ch = t >> 5;
    float acc = 0.0f;
    for (int j = ch * 128; j < (ch + 1) * 128; ++j)
        acc += s[j] * base[(long long)j * 3 * ND + 2 * ND + h * NHD + i];
    ored[ch][i] = acc;
    __syncthreads();
    if (t < NHD) {
        const float r = ored[0][t] + ored[1][t] + ored[2][t] + ored[3][t];
        o[((long long)b * NN + n) * ND + h * NHD + t] = r * inv;
    }
}

__global__ void score_kernel(const float* __restrict__ x,
                             const float* __restrict__ sw,
                             const float* __restrict__ sb,
                             float* __restrict__ scores) {
    const int i = blockIdx.x * blockDim.x + threadIdx.x;
    if (i >= NB * NN) return;
    const float* row = x + (long long)i * ND;
    float acc = sb[0];
#pragma unroll 8
    for (int d = 0; d < ND; ++d) acc += row[d] * sw[d];
    scores[i] = acc;
}

__global__ __launch_bounds__(512) void topk_kernel(
    const float* __restrict__ scores, int* __restrict__ topidx) {
    const int b = blockIdx.x, t = threadIdx.x;
    __shared__ float v[NN];
    __shared__ int   ix[NN];
    v[t]  = scores[b * NN + t];
    ix[t] = t;
    __syncthreads();
    for (int k = 2; k <= NN; k <<= 1) {
        for (int j = k >> 1; j > 0; j >>= 1) {
            const int p = t ^ j;
            if (p > t) {
                const bool desc = ((t & k) == 0);
                const float va = v[t], vb = v[p];
                const int ia = ix[t], ib = ix[p];
                const bool a_first = (va > vb) || (va == vb && ia < ib);
                if (desc ? !a_first : a_first) {
                    v[t] = vb; v[p] = va; ix[t] = ib; ix[p] = ia;
                }
            }
            __syncthreads();
        }
    }
    if (t < NK) topidx[b * NK + t] = ix[t];
}

__global__ void gather_topk(const float* __restrict__ x,
                            const int* __restrict__ topidx,
                            float* __restrict__ xtop) {
    const int bk = blockIdx.x;
    const int d  = threadIdx.x;
    const int b = bk / NK, k = bk % NK;
    const int n = topidx[b * NK + k];
    xtop[(long long)bk * ND + d] = x[((long long)b * NN + n) * ND + d];
}

__global__ __launch_bounds__(256) void pool_kernel(
    const float* __restrict__ xtop, const float* __restrict__ aw,
    const float* __restrict__ ab, float* __restrict__ pooled) {
    const int b = blockIdx.x, t = threadIdx.x;
    __shared__ float a[NK];
    __shared__ float red[256];
    const float* xb = xtop + (long long)b * NK * ND;
    float acc = ab[0];
    const float* row = xb + (long long)t * ND;
#pragma unroll 8
    for (int d = 0; d < ND; ++d) acc += row[d] * aw[d];
    a[t] = acc;
    red[t] = acc;
    __syncthreads();
    for (int s = 128; s > 0; s >>= 1) {
        if (t < s) red[t] = fmaxf(red[t], red[t + s]);
        __syncthreads();
    }
    const float m = red[0];
    __syncthreads();
    const float e = __expf(a[t] - m);
    a[t] = e;
    red[t] = e;
    __syncthreads();
    for (int s = 128; s > 0; s >>= 1) {
        if (t < s) red[t] += red[t + s];
        __syncthreads();
    }
    const float inv = 1.0f / red[0];
    __syncthreads();
    float p = 0.0f;
    for (int k = 0; k < NK; ++k) p += a[k] * xb[(long long)k * ND + t];
    pooled[(long long)b * ND + t] = p * inv;
}

// ---------------------------------------------------------------------------
// Host orchestration
// ---------------------------------------------------------------------------
static inline void launch_gemm(hipStream_t s, const _Float16* A,
                               const _Float16* W, const float* bias, float* C,
                               int M, int Nc, int Kd, int lda, int ldw,
                               int ldc, int relu) {
    dim3 grid(M / 128, Nc / 16, 1);
    wmma_gemm_f16t<<<grid, 128, 0, s>>>(A, W, bias, C, M, Nc, Kd, lda, ldw,
                                        ldc, relu);
}

static inline void launch_cvt(hipStream_t s, const float* in, _Float16* out,
                              long long n) {
    cvt_f32_f16<<<(int)((n + 255) / 256), 256, 0, s>>>(in, out, n);
}

extern "C" void kernel_launch(void* const* d_in, const int* in_sizes, int n_in,
                              void* d_out, int out_size, void* d_ws,
                              size_t ws_size, hipStream_t stream) {
    (void)in_sizes; (void)n_in; (void)out_size; (void)ws_size;

    const int*   ids = (const int*)d_in[0];        // (B,N)
    const int*   ei  = (const int*)d_in[1];        // (B,2,E)
    const int*   et  = (const int*)d_in[2];        // (B,E)
    const float* emb = (const float*)d_in[3];      // (V,D)
    auto P = [&](int l, int p) -> const float* {
        return (const float*)d_in[4 + 19 * l + p];
    };
    const float* score_w = (const float*)d_in[4 + 19 * NL + 0];
    const float* score_b = (const float*)d_in[4 + 19 * NL + 1];
    const float* attn_w  = (const float*)d_in[4 + 19 * NL + 2];
    const float* attn_b  = (const float*)d_in[4 + 19 * NL + 3];

    // workspace layout
    const long long SZ = (long long)NB * NN * ND;          // 4,194,304
    float* xbuf = (float*)d_ws;                            // x / x1
    float* zbuf = xbuf + SZ;                               // z
    float* tbuf = zbuf + SZ;                               // local / o / ff / t
    float* rbuf = tbuf + SZ;                               // rgcn hr / attn out
    float* qkv  = rbuf + SZ;                               // (B,N,3D)
    float* hbuf = qkv + 3 * SZ;                            // (N,DFF) per-batch
    float* cnt  = hbuf + (long long)NN * NDFF;             // (B,N,R)
    float* scor = cnt + (long long)NB * NN * NR;           // (B,N)
    int*   tidx = (int*)(scor + (long long)NB * NN);       // (B,K)
    // f16 mirrors
    _Float16* a16 = (_Float16*)(tidx + (long long)NB * NK);        // SZ halfs
    _Float16* h16 = a16 + SZ;                                      // N*DFF halfs
    _Float16* w_rgcn = h16 + (long long)NN * NDFF;                 // R*D*D
    _Float16* w_root = w_rgcn + (long long)NR * ND * ND;           // D*D
    _Float16* w_in   = w_root + (long long)ND * ND;                // 3D*D
    _Float16* w_out  = w_in + (long long)3 * ND * ND;              // D*D
    _Float16* w_l1   = w_out + (long long)ND * ND;                 // DFF*D
    _Float16* w_l2   = w_l1 + (long long)NDFF * ND;                // D*DFF

    float* out_pooled = (float*)d_out;                     // (B,1,D)
    float* out_xtop   = out_pooled + (long long)NB * ND;   // (B,K,D)

    embed_kernel<<<NB * NN, ND, 0, stream>>>(ids, emb, xbuf);

    for (int l = 0; l < NL; ++l) {
        const float* rgcn_w    = P(l, 0);
        const float* rgcn_root = P(l, 1);
        const float* rgcn_b    = P(l, 2);
        const float* in_w      = P(l, 3);
        const float* in_b      = P(l, 4);
        const float* out_w     = P(l, 5);
        const float* out_b     = P(l, 6);
        const float* l1_w      = P(l, 7);
        const float* l1_b      = P(l, 8);
        const float* l2_w      = P(l, 9);
        const float* l2_b      = P(l, 10);
        const float* tn1w = P(l, 11); const float* tn1b = P(l, 12);
        const float* tn2w = P(l, 13); const float* tn2b = P(l, 14);
        const float* n1w  = P(l, 15); const float* n1b  = P(l, 16);
        const float* n2w  = P(l, 17); const float* n2b  = P(l, 18);

        // ---- weight conversion (f32 -> f16; x@W forms pre-transposed)
        {
            const int nwt = ND * ND;
            for (int r = 0; r < NR; ++r)
                cvt_f32_f16_T<<<(nwt + 255) / 256, 256, 0, stream>>>(
                    rgcn_w + (long long)r * nwt, w_rgcn + (long long)r * nwt,
                    ND, ND);
            cvt_f32_f16_T<<<(nwt + 255) / 256, 256, 0, stream>>>(
                rgcn_root, w_root, ND, ND);
            launch_cvt(stream, in_w, w_in, (long long)3 * ND * ND);
            launch_cvt(stream, out_w, w_out, (long long)ND * ND);
            launch_cvt(stream, l1_w, w_l1, (long long)NDFF * ND);
            launch_cvt(stream, l2_w, w_l2, (long long)ND * NDFF);
        }

        // ---- RGCN degree counts
        {
            const long long ncnt = (long long)NB * NN * NR;
            zero_kernel<<<(int)((ncnt + 255) / 256), 256, 0, stream>>>(cnt, ncnt);
            rgcn_count<<<(NB * NE + 255) / 256, 256, 0, stream>>>(ei, et, cnt);
        }

        // ---- RGCN: local = x@root + b ; per-rel hr_r = x@W_r ; scatter
        launch_cvt(stream, xbuf, a16, SZ);          // one cvt feeds 17 GEMMs
        launch_gemm(stream, a16, w_root, rgcn_b, tbuf,
                    NB * NN, ND, ND, ND, ND, ND, 0);
        for (int r = 0; r < NR; ++r) {
            launch_gemm(stream, a16, w_rgcn + (long long)r * ND * ND, nullptr,
                        rbuf, NB * NN, ND, ND, ND, ND, ND, 0);
            rgcn_scatter<<<dim3(NE, NB), 256, 0, stream>>>(ei, et, cnt, rbuf,
                                                           tbuf, r);
        }
        ln_add<<<NB * NN, 256, 0, stream>>>(xbuf, tbuf, n1w, n1b, xbuf);

        // ---- attention
        launch_cvt(stream, xbuf, a16, SZ);
        launch_gemm(stream, a16, w_in, in_b, qkv,
                    NB * NN, 3 * ND, ND, ND, ND, 3 * ND, 0);
        attn_fused<<<dim3(NN, NH, NB), 128, 0, stream>>>(qkv, rbuf);
        launch_cvt(stream, rbuf, a16, SZ);
        launch_gemm(stream, a16, w_out, out_b, tbuf,
                    NB * NN, ND, ND, ND, ND, ND, 0);
        ln_add<<<NB * NN, 256, 0, stream>>>(xbuf, tbuf, tn1w, tn1b, zbuf);

        // ---- FFN (per batch)
        launch_cvt(stream, zbuf, a16, SZ);
        for (int b = 0; b < NB; ++b) {
            launch_gemm(stream, a16 + (long long)b * NN * ND, w_l1, l1_b, hbuf,
                        NN, NDFF, ND, ND, ND, NDFF, /*relu=*/1);
            launch_cvt(stream, hbuf, h16, (long long)NN * NDFF);
            launch_gemm(stream, h16, w_l2, l2_b,
                        tbuf + (long long)b * NN * ND,
                        NN, ND, NDFF, NDFF, NDFF, ND, 0);
        }
        ln_add<<<NB * NN, 256, 0, stream>>>(zbuf, tbuf, tn2w, tn2b, tbuf);
        ln_add<<<NB * NN, 256, 0, stream>>>(xbuf, tbuf, n2w, n2b, xbuf);
    }

    // ---- scoring, top-k, pooling
    score_kernel<<<(NB * NN + 255) / 256, 256, 0, stream>>>(xbuf, score_w,
                                                            score_b, scor);
    topk_kernel<<<NB, NN, 0, stream>>>(scor, tidx);
    gather_topk<<<NB * NK, ND, 0, stream>>>(xbuf, tidx, out_xtop);
    pool_kernel<<<NB, NK, 0, stream>>>(out_xtop, attn_w, attn_b, out_pooled);
}
